// SelfAttention_30021821399257
// MI455X (gfx1250) — matrix operand order repaired and checked by reading, not compiled, not run on hardware
//
#include <hip/hip_runtime.h>

// ---- Problem constants (match reference) ----
#define NB    4
#define CIN   256
#define HWPIX 16384      // 128*128
#define NG    32         // group-norm groups
#define NHEAD 64
// qkv channels = 768; head h occupies channels [h*12, h*12+12): q=0..3, k=4..7, v=8..11

// ---- CDNA5 WMMA types ----
typedef __bf16 bf16;
typedef __attribute__((ext_vector_type(16))) bf16  v16bf;
typedef __attribute__((ext_vector_type(8)))  float v8f;
typedef __attribute__((ext_vector_type(4)))  unsigned int u32x4;
typedef __attribute__((ext_vector_type(2)))  unsigned int u32x2;

union Frag { u32x4 q[2]; v16bf v; };

__device__ __forceinline__ unsigned short f2bf(float f) {
  unsigned int u = __float_as_uint(f);
  u += 0x7FFFu + ((u >> 16) & 1u);          // round-to-nearest-even
  return (unsigned short)(u >> 16);
}
__device__ __forceinline__ float bf2f(unsigned short h) {
  return __uint_as_float(((unsigned int)h) << 16);
}

// Low 32 bits of a flat shared-memory pointer == LDS byte offset (hardware adds
// LDS_BASE; flat aperture keeps the offset in addr[31:0], ISA 10.2).
__device__ __forceinline__ unsigned int lds_off(const void* p) {
  return (unsigned int)(unsigned long long)p;
}
// CDNA5 async global->LDS copy, 16B per lane, tracked by ASYNCcnt (ISA 10.x /
// CDNA5 ISA). VDST vgpr carries the per-lane LDS byte address.
__device__ __forceinline__ void async_copy_b128(unsigned int lds_byte,
                                                const unsigned short* g) {
  asm volatile("global_load_async_to_lds_b128 %0, %1, off"
               :: "v"(lds_byte), "v"(g) : "memory");
}
__device__ __forceinline__ void wait_async0() {
  asm volatile("s_wait_asynccnt 0x0" ::: "memory");
}

// A-fragment: 16x32 bf16, weights row-major (M x K). Lane L: row m0+(L&15),
// sel=L>>4; j0..7 <- K[k0+8*sel .. +7], j8..15 <- K[k0+16+8*sel .. +7].
__device__ __forceinline__ void load_afrag(Frag& a, const unsigned short* W,
                                           int ldk, int m0, int k0, int lane) {
  const unsigned short* p = W + (size_t)(m0 + (lane & 15)) * ldk + k0 + 8 * (lane >> 4);
  a.q[0] = *reinterpret_cast<const u32x4*>(p);
  a.q[1] = *reinterpret_cast<const u32x4*>(p + 16);
}
// B-fragment: 32x16 bf16 read from padded LDS tile (row stride LDSPITCH halves).
// Lane L: column base+(L&15), j0..7 <- K[k0+8*sel .. +7], j8..15 <- K[k0+16+8*sel .. +7].
#define LDSPITCH 264   // 256 + 8 halves pad: dword stride 132 = 4 mod 64 banks
__device__ __forceinline__ void load_bfrag_lds(Frag& b, const unsigned short* sB,
                                               int nloc, int k0, int lane) {
  const unsigned short* p = sB + (nloc + (lane & 15)) * LDSPITCH + k0 + 8 * (lane >> 4);
  b.q[0] = *reinterpret_cast<const u32x4*>(p);
  b.q[1] = *reinterpret_cast<const u32x4*>(p + 16);
}

// ---- 1) weight f32 -> bf16 (row-major kept; A-frag loads handle layout) ----
__global__ void wconv_kernel(const float* __restrict__ wq, const float* __restrict__ wo,
                             unsigned short* __restrict__ wqb, unsigned short* __restrict__ wob) {
  int i = blockIdx.x * 256 + threadIdx.x;
  if (i < 768 * 256) wqb[i] = f2bf(wq[i]);
  if (i < 256 * 256) wob[i] = f2bf(wo[i]);
}

// ---- 2) group-norm statistics: one block per (b,g), unbiased variance ----
__global__ void gn_stats_kernel(const float* __restrict__ x, float* __restrict__ stats) {
  const int bg = blockIdx.x;                       // b*NG+g ; channels contiguous
  const int tid = threadIdx.x;
  const size_t base = (size_t)bg * (CIN / NG) * HWPIX;
  const int N = (CIN / NG) * HWPIX;                // 131072
  float s = 0.f, ss = 0.f;
  for (int i = tid; i < N; i += 256) {
    float v = x[base + i];
    s += v; ss += v * v;
  }
  __shared__ float rs[256], rss[256];
  rs[tid] = s; rss[tid] = ss;
  __syncthreads();
  for (int st = 128; st > 0; st >>= 1) {
    if (tid < st) { rs[tid] += rs[tid + st]; rss[tid] += rss[tid + st]; }
    __syncthreads();
  }
  if (tid == 0) {
    float sum = rs[0], sumsq = rss[0];
    float mean = sum / (float)N;
    float var  = (sumsq - sum * mean) / (float)(N - 1);   // ddof=1
    stats[2 * bg]     = mean;
    stats[2 * bg + 1] = rsqrtf(var + 1e-5f);
  }
}

// ---- 3) normalize + transpose to [b][pixel][channel] bf16 (WMMA-B layout) ----
__global__ void gn_norm_kernel(const float* __restrict__ x,
                               const float* __restrict__ gamma,
                               const float* __restrict__ beta,
                               const float* __restrict__ stats,
                               unsigned short* __restrict__ hbf) {
  __shared__ float tile[16][17];
  const int b  = blockIdx.z;
  const int c0 = blockIdx.y * 16;
  const int n0 = blockIdx.x * 16;
  const int tx = threadIdx.x, ty = threadIdx.y;
  const int c = c0 + ty, n = n0 + tx;
  const int g = c >> 3;
  const float mean = stats[2 * (b * NG + g)];
  const float rstd = stats[2 * (b * NG + g) + 1];
  const float v = x[((size_t)b * CIN + c) * HWPIX + n];
  tile[ty][tx] = (v - mean) * rstd * gamma[c] + beta[c];
  __syncthreads();
  hbf[((size_t)b * HWPIX + (n0 + ty)) * CIN + (c0 + tx)] = f2bf(tile[tx][ty]);
}

// ---- 4) QKV GEMM: (768x256) @ h(256 x HW). Block = 8 waves = 128 rows x 64
// pixels; the 32KB B tile is async-staged into padded LDS once and shared by
// all 8 M-tiles (8x cut in L2 B-traffic). ----
__global__ void qkv_gemm_kernel(const unsigned short* __restrict__ Wq,   // 768x256
                                const unsigned short* __restrict__ Hbf,  // (B,HW,256)
                                const float* __restrict__ bias,          // 768
                                unsigned short* __restrict__ QKV) {      // (B,768,HW) bf16
  __shared__ unsigned short sB[64 * LDSPITCH];
  const int tid  = threadIdx.x;
  const int lane = tid & 31;
  const int wave = tid >> 5;
  const int nt = blockIdx.x & 255;               // 256 tiles of 64 pixels
  const int mb = (blockIdx.x >> 8) % 6;          // 6 blocks of 128 out-channels
  const int b  = blockIdx.x / (256 * 6);
  const int n0 = nt * 64;
  const int m0 = mb * 128 + wave * 16;

  { // stage B: 64 rows x 512B (contiguous in memory) -> padded LDS rows
    const int r = tid >> 2, qd = tid & 3;
    const unsigned short* src = Hbf + ((size_t)b * HWPIX + n0 + r) * CIN + qd * 64;
    const unsigned int dst = lds_off(&sB[r * LDSPITCH + qd * 64]);
#pragma unroll
    for (int i = 0; i < 8; ++i) async_copy_b128(dst + i * 16, src + i * 8);
  }
  wait_async0();
  __syncthreads();

  v8f acc[4] = {};
  for (int k0 = 0; k0 < CIN; k0 += 32) {
    Frag a; load_afrag(a, Wq, CIN, m0, k0, lane);
#pragma unroll
    for (int t = 0; t < 4; ++t) {
      Frag bq; load_bfrag_lds(bq, sB, t * 16, k0, lane);
      acc[t] = __builtin_amdgcn_wmma_f32_16x16x32_bf16(
          false, a.v, false, bq.v, (short)0, acc[t], false, false);
    }
  }
#pragma unroll
  for (int t = 0; t < 4; ++t) {
    const int n = n0 + t * 16 + (lane & 15);
#pragma unroll
    for (int r = 0; r < 8; ++r) {
      const int m = m0 + r + 8 * (lane >> 4);
      QKV[((size_t)b * 768 + m) * HWPIX + n] = f2bf(acc[t][r] + bias[m]);
    }
  }
}

// ---- 5) scores[b,h,d,e] = (q.k)/sqrt(4): 16 dots of length HW per (b,h) ----
__global__ void scores_kernel(const unsigned short* __restrict__ QKV,
                              float* __restrict__ scores) {
  const int bh = blockIdx.x;             // b*64+h
  const int tid = threadIdx.x;
  const unsigned short* qb = QKV + ((size_t)(bh >> 6) * 768 + (bh & 63) * 12) * HWPIX;
  const unsigned short* kb = qb + (size_t)4 * HWPIX;
  float acc[16];
#pragma unroll
  for (int i = 0; i < 16; ++i) acc[i] = 0.f;
  for (int n = tid; n < HWPIX; n += 256) {
    float qv[4], kv[4];
#pragma unroll
    for (int d = 0; d < 4; ++d) qv[d] = bf2f(qb[(size_t)d * HWPIX + n]);
#pragma unroll
    for (int e = 0; e < 4; ++e) kv[e] = bf2f(kb[(size_t)e * HWPIX + n]);
#pragma unroll
    for (int d = 0; d < 4; ++d)
#pragma unroll
      for (int e = 0; e < 4; ++e) acc[d * 4 + e] += qv[d] * kv[e];
  }
  __shared__ float red[16 * 256];
#pragma unroll
  for (int i = 0; i < 16; ++i) red[i * 256 + tid] = acc[i];
  __syncthreads();
  for (int st = 128; st > 0; st >>= 1) {
    if (tid < st) {
#pragma unroll
      for (int i = 0; i < 16; ++i) red[i * 256 + tid] += red[i * 256 + tid + st];
    }
    __syncthreads();
  }
  if (tid < 16) scores[bh * 16 + tid] = red[tid * 256] * 0.5f;   // 1/sqrt(hd)
}

// ---- 6) softmax over HEADS axis (axis=1), per (b,d,e) ----
__global__ void softmax_kernel(float* __restrict__ scores) {
  const int tid = threadIdx.x;
  if (tid >= NB * 16) return;
  const int b = tid >> 4, de = tid & 15;
  float m = -3.4e38f;
  for (int h = 0; h < NHEAD; ++h) m = fmaxf(m, scores[(b * NHEAD + h) * 16 + de]);
  float s = 0.f;
  for (int h = 0; h < NHEAD; ++h) s += __expf(scores[(b * NHEAD + h) * 16 + de] - m);
  const float inv = 1.f / s;
  for (int h = 0; h < NHEAD; ++h) {
    const int i = (b * NHEAD + h) * 16 + de;
    scores[i] = __expf(scores[i] - m) * inv;
  }
}

// ---- 7) attn[b][n][c=h*4+d] = sum_e s[b,h,d,e]*v[b,h,e,n], write bf16 B-layout ----
__global__ void attn_kernel(const unsigned short* __restrict__ QKV,
                            const float* __restrict__ scores,
                            unsigned short* __restrict__ Abf) {
  __shared__ float s_s[NHEAD * 16];
  const int b = blockIdx.x >> 8;
  const int chunk = blockIdx.x & 255;
  const int tid = threadIdx.x;
  for (int i = tid; i < NHEAD * 16; i += 256) s_s[i] = scores[b * NHEAD * 16 + i];
  __syncthreads();
  const int n = chunk * 64 + (tid & 63);
  for (int h = (tid >> 6); h < NHEAD; h += 4) {
    const unsigned short* vb = QKV + ((size_t)b * 768 + h * 12 + 8) * HWPIX + n;
    float vv[4];
#pragma unroll
    for (int e = 0; e < 4; ++e) vv[e] = bf2f(vb[(size_t)e * HWPIX]);
    float o0 = 0.f, o1 = 0.f, o2 = 0.f, o3 = 0.f;
#pragma unroll
    for (int e = 0; e < 4; ++e) {
      o0 += s_s[h * 16 + 0 + e] * vv[e];
      o1 += s_s[h * 16 + 4 + e] * vv[e];
      o2 += s_s[h * 16 + 8 + e] * vv[e];
      o3 += s_s[h * 16 + 12 + e] * vv[e];
    }
    u32x2 pk;
    pk.x = (unsigned int)f2bf(o0) | ((unsigned int)f2bf(o1) << 16);
    pk.y = (unsigned int)f2bf(o2) | ((unsigned int)f2bf(o3) << 16);
    *reinterpret_cast<u32x2*>(Abf + ((size_t)b * HWPIX + n) * CIN + h * 4) = pk;
  }
}

// ---- 8) out GEMM: (256x256) @ attn + bias + skip; same LDS-staged scheme ----
__global__ void out_gemm_kernel(const unsigned short* __restrict__ Wo,   // 256x256
                                const unsigned short* __restrict__ Abf,  // (B,HW,256)
                                const float* __restrict__ bias,
                                const float* __restrict__ skip,
                                float* __restrict__ out) {
  __shared__ unsigned short sB[64 * LDSPITCH];
  const int tid  = threadIdx.x;
  const int lane = tid & 31;
  const int wave = tid >> 5;
  const int nt = blockIdx.x & 255;
  const int mb = (blockIdx.x >> 8) & 1;          // 2 blocks of 128 out-channels
  const int b  = blockIdx.x >> 9;
  const int n0 = nt * 64;
  const int m0 = mb * 128 + wave * 16;

  {
    const int r = tid >> 2, qd = tid & 3;
    const unsigned short* src = Abf + ((size_t)b * HWPIX + n0 + r) * CIN + qd * 64;
    const unsigned int dst = lds_off(&sB[r * LDSPITCH + qd * 64]);
#pragma unroll
    for (int i = 0; i < 8; ++i) async_copy_b128(dst + i * 16, src + i * 8);
  }
  wait_async0();
  __syncthreads();

  v8f acc[4] = {};
  for (int k0 = 0; k0 < CIN; k0 += 32) {
    Frag a; load_afrag(a, Wo, CIN, m0, k0, lane);
#pragma unroll
    for (int t = 0; t < 4; ++t) {
      Frag bq; load_bfrag_lds(bq, sB, t * 16, k0, lane);
      acc[t] = __builtin_amdgcn_wmma_f32_16x16x32_bf16(
          false, a.v, false, bq.v, (short)0, acc[t], false, false);
    }
  }
#pragma unroll
  for (int t = 0; t < 4; ++t) {
    const int n = n0 + t * 16 + (lane & 15);
#pragma unroll
    for (int r = 0; r < 8; ++r) {
      const int m = m0 + r + 8 * (lane >> 4);
      const size_t idx = ((size_t)b * CIN + m) * HWPIX + n;
      out[idx] = acc[t][r] + bias[m] + skip[idx];
    }
  }
}

extern "C" void kernel_launch(void* const* d_in, const int* in_sizes, int n_in,
                              void* d_out, int out_size, void* d_ws, size_t ws_size,
                              hipStream_t stream) {
  const float* x     = (const float*)d_in[0];
  const float* gnw   = (const float*)d_in[1];
  const float* gnb   = (const float*)d_in[2];
  const float* qkv_w = (const float*)d_in[3];
  const float* qkv_b = (const float*)d_in[4];
  const float* out_w = (const float*)d_in[5];
  const float* out_b = (const float*)d_in[6];
  float* out = (float*)d_out;

  // Workspace layout (~129 MB; qkv+attn sized to stay inside 192 MB L2):
  char* ws = (char*)d_ws;
  float*          stats  = (float*)(ws);                              // 1 KB
  float*          scores = (float*)(ws + 4096);                       // 16 KB
  unsigned short* wq_bf  = (unsigned short*)(ws + 32768);             // 384 KB
  unsigned short* wo_bf  = (unsigned short*)(ws + 32768 + 768*256*2); // 128 KB
  unsigned short* h_bf   = (unsigned short*)(ws + (1 << 20));         // 32 MB
  unsigned short* qkv_bf = (unsigned short*)(ws + (1 << 20)
                           + (size_t)NB * HWPIX * CIN * 2);           // 96 MB
  unsigned short* attn_bf = h_bf;   // h dead after QKV GEMM -> reuse buffer

  wconv_kernel   <<<768, 256, 0, stream>>>(qkv_w, out_w, wq_bf, wo_bf);
  gn_stats_kernel<<<NB * NG, 256, 0, stream>>>(x, stats);
  gn_norm_kernel <<<dim3(HWPIX / 16, CIN / 16, NB), dim3(16, 16), 0, stream>>>(
      x, gnw, gnb, stats, h_bf);
  qkv_gemm_kernel<<<NB * 6 * 256, 256, 0, stream>>>(wq_bf, h_bf, qkv_b, qkv_bf);
  scores_kernel  <<<NB * NHEAD, 256, 0, stream>>>(qkv_bf, scores);
  softmax_kernel <<<1, 64, 0, stream>>>(scores);
  attn_kernel    <<<NB * (HWPIX / 64), 256, 0, stream>>>(qkv_bf, scores, attn_bf);
  out_gemm_kernel<<<NB * 2 * 256, 256, 0, stream>>>(wo_bf, attn_bf, out_b, x, out);
}
